// GraphSAGE_89876485636273
// MI455X (gfx1250) — compile-verified, matrix operand order checked by reading, not run on hardware
//
#include <hip/hip_runtime.h>
#include <hip/hip_bf16.h>

#define N_NODES 100000
#define N_EDGES 3200000

typedef __attribute__((ext_vector_type(16))) __bf16 v16bf;
typedef __attribute__((ext_vector_type(8)))  float  v8f;

// ---------------------------------------------------------------------------
// Utility: zero a float buffer (graph-capture safe, no memset)
// ---------------------------------------------------------------------------
__global__ void zero_kernel(float* __restrict__ p, long n) {
    long i = (long)blockIdx.x * blockDim.x + threadIdx.x;
    long stride = (long)gridDim.x * blockDim.x;
    for (; i < n; i += stride) p[i] = 0.0f;
}

// ---------------------------------------------------------------------------
// In-degree count (computed once; same graph for all 3 layers)
// ---------------------------------------------------------------------------
__global__ void count_kernel(const int* __restrict__ dst, float* __restrict__ cnt, int E) {
    int t = blockIdx.x * blockDim.x + threadIdx.x;
    if (t >= E) return;
    __hip_atomic_fetch_add(&cnt[dst[t]], 1.0f, __ATOMIC_RELAXED, __HIP_MEMORY_SCOPE_AGENT);
}

// ---------------------------------------------------------------------------
// Edge scatter: s[dst] += feat[src]  (one thread per (edge, 4-channel group))
// float4 gather is coalesced; adds are global_atomic_add_f32
// ---------------------------------------------------------------------------
__global__ void edge_scatter_kernel(const float* __restrict__ feat,
                                    const int* __restrict__ src,
                                    const int* __restrict__ dst,
                                    float* __restrict__ s,
                                    long total, int Cg, int C) {
    long t = (long)blockIdx.x * blockDim.x + threadIdx.x;
    if (t >= total) return;
    long e = t / Cg;
    int  g = (int)(t - e * Cg) * 4;
    int  si = src[e], di = dst[e];
    float4 v = *(const float4*)(feat + (size_t)si * C + g);
    float* p = s + (size_t)di * C + g;
    __hip_atomic_fetch_add(p + 0, v.x, __ATOMIC_RELAXED, __HIP_MEMORY_SCOPE_AGENT);
    __hip_atomic_fetch_add(p + 1, v.y, __ATOMIC_RELAXED, __HIP_MEMORY_SCOPE_AGENT);
    __hip_atomic_fetch_add(p + 2, v.z, __ATOMIC_RELAXED, __HIP_MEMORY_SCOPE_AGENT);
    __hip_atomic_fetch_add(p + 3, v.w, __ATOMIC_RELAXED, __HIP_MEMORY_SCOPE_AGENT);
}

// ---------------------------------------------------------------------------
// Pack f32 row-major weight W[K x N] into WMMA B-operand bf16 lane layout:
// tile (kt, nt) -> 32 lanes x 16 bf16. Lane holds column n = nt*16+(lane&15),
// khalf = (lane>>4)*8, vgpr-pair r: K = kt*32 + khalf + (r<4 ? 2r : 16+2(r-4)) + b
// ---------------------------------------------------------------------------
__global__ void pack_weights_kernel(const float* __restrict__ W, __bf16* __restrict__ Wp,
                                    int K, int N) {
    int total = K * N; // == Ktiles*Ntiles*512
    int t = blockIdx.x * blockDim.x + threadIdx.x;
    if (t >= total) return;
    int Ntiles = N >> 4;
    int j    = t & 15;
    int lane = (t >> 4) & 31;
    int tile = t >> 9;
    int nt = tile % Ntiles;
    int kt = tile / Ntiles;
    int n  = nt * 16 + (lane & 15);
    int khalf = (lane >> 4) << 3;
    int r = j >> 1, b = j & 1;
    int k = kt * 32 + khalf + ((r < 4) ? (2 * r) : (16 + 2 * (r - 4))) + b;
    Wp[t] = (__bf16)W[(size_t)k * N + n];
}

// Build WMMA A-operand (16 bf16) for one lane from f32 row data, scaled.
// K offsets covered: base+[0..7] and base+[16..23]  (per ISA 16-bit A layout)
__device__ inline v16bf make_a_bf16(const float* __restrict__ p, float scale) {
    float4 f0 = *(const float4*)(p);
    float4 f1 = *(const float4*)(p + 4);
    float4 f2 = *(const float4*)(p + 16);
    float4 f3 = *(const float4*)(p + 20);
    v16bf a;
    a[0]  = (__bf16)(f0.x * scale); a[1]  = (__bf16)(f0.y * scale);
    a[2]  = (__bf16)(f0.z * scale); a[3]  = (__bf16)(f0.w * scale);
    a[4]  = (__bf16)(f1.x * scale); a[5]  = (__bf16)(f1.y * scale);
    a[6]  = (__bf16)(f1.z * scale); a[7]  = (__bf16)(f1.w * scale);
    a[8]  = (__bf16)(f2.x * scale); a[9]  = (__bf16)(f2.y * scale);
    a[10] = (__bf16)(f2.z * scale); a[11] = (__bf16)(f2.w * scale);
    a[12] = (__bf16)(f3.x * scale); a[13] = (__bf16)(f3.y * scale);
    a[14] = (__bf16)(f3.z * scale); a[15] = (__bf16)(f3.w * scale);
    return a;
}

// ---------------------------------------------------------------------------
// SAGE dense step via WMMA: out = relu(agg @ Wl + x @ Wr + bias)
// agg = s/max(cnt,1) formed on the fly. One wave (32 threads) per 16x16 tile.
// grid = (N_NODES/16, Nout/16), block = 32.  EXEC stays all-ones (wmma req).
// ---------------------------------------------------------------------------
__global__ void sage_gemm_kernel(const float* __restrict__ s,
                                 const float* __restrict__ cnt,
                                 const float* __restrict__ xin,
                                 const __bf16* __restrict__ WlP,
                                 const __bf16* __restrict__ WrP,
                                 const float* __restrict__ bias,
                                 float* __restrict__ out,
                                 int K, int Nout, int Ntiles) {
    const int lane  = threadIdx.x;          // 0..31
    const int m0    = blockIdx.x << 4;      // node tile
    const int nt    = blockIdx.y;           // column tile
    const int row   = m0 + (lane & 15);
    const int khalf = (lane >> 4) << 3;     // 0 or 8

    const float invc = 1.0f / fmaxf(cnt[row], 1.0f);

    v8f acc = {};
    const int Ktiles = K >> 5;
    for (int kt = 0; kt < Ktiles; ++kt) {
        const int kbase = (kt << 5) + khalf;
        // agg @ Wl
        v16bf aA = make_a_bf16(s + (size_t)row * K + kbase, invc);
        v16bf bL = *(const v16bf*)(WlP + (((size_t)(kt * Ntiles + nt)) << 9) + (lane << 4));
        acc = __builtin_amdgcn_wmma_f32_16x16x32_bf16(false, aA, false, bL,
                                                      (short)0, acc, false, false);
        // x @ Wr
        v16bf aX = make_a_bf16(xin + (size_t)row * K + kbase, 1.0f);
        v16bf bR = *(const v16bf*)(WrP + (((size_t)(kt * Ntiles + nt)) << 9) + (lane << 4));
        acc = __builtin_amdgcn_wmma_f32_16x16x32_bf16(false, aX, false, bR,
                                                      (short)0, acc, false, false);
    }

    // C/D layout: vgpr v holds M = v + 8*(lane>=16), N = lane&15
    const int col   = (nt << 4) + (lane & 15);
    const float bv  = bias[col];
    const int rbase = m0 + ((lane >> 4) << 3);
#pragma unroll
    for (int v = 0; v < 8; ++v) {
        float r = acc[v] + bv;
        r = fmaxf(r, 0.0f);                  // ReLU (all 3 SAGE layers)
        out[(size_t)(rbase + v) * Nout + col] = r;
    }
}

// ---------------------------------------------------------------------------
// Final head: out[i] = dot(h2[i, 0:16], Wc) + bc   (no relu)
// ---------------------------------------------------------------------------
__global__ void final_head_kernel(const float* __restrict__ h2,
                                  const float* __restrict__ Wc,
                                  const float* __restrict__ bc,
                                  float* __restrict__ out, int Nn) {
    int i = blockIdx.x * blockDim.x + threadIdx.x;
    if (i >= Nn) return;
    const float* p = h2 + (size_t)i * 16;
    float acc = bc[0];
#pragma unroll
    for (int c = 0; c < 16; c += 4) {
        float4 h = *(const float4*)(p + c);
        float4 w = *(const float4*)(Wc + c);
        acc += h.x * w.x + h.y * w.y + h.z * w.z + h.w * w.w;
    }
    out[i] = acc;
}

// ---------------------------------------------------------------------------
extern "C" void kernel_launch(void* const* d_in, const int* in_sizes, int n_in,
                              void* d_out, int out_size, void* d_ws, size_t ws_size,
                              hipStream_t stream) {
    const float* x   = (const float*)d_in[0];
    const int*   ei  = (const int*)d_in[1];
    const float* Wl0 = (const float*)d_in[2];
    const float* Wr0 = (const float*)d_in[3];
    const float* b0  = (const float*)d_in[4];
    const float* Wl1 = (const float*)d_in[5];
    const float* Wr1 = (const float*)d_in[6];
    const float* b1  = (const float*)d_in[7];
    const float* Wl2 = (const float*)d_in[8];
    const float* Wr2 = (const float*)d_in[9];
    const float* b2  = (const float*)d_in[10];
    const float* Wc  = (const float*)d_in[11];
    const float* bc  = (const float*)d_in[12];

    const int* src = ei;
    const int* dst = ei + N_EDGES;

    // Workspace carve-up (all offsets 64B aligned)
    char* ws = (char*)d_ws;
    float* cnt = (float*)ws;                  ws += (size_t)N_NODES * 4;
    float* s   = (float*)ws;                  ws += (size_t)N_NODES * 64 * 4;
    float* h0  = (float*)ws;                  ws += (size_t)N_NODES * 64 * 4;
    float* h1  = (float*)ws;                  ws += (size_t)N_NODES * 32 * 4;
    float* h2  = (float*)ws;                  ws += (size_t)N_NODES * 16 * 4;
    __bf16* Wl0p = (__bf16*)ws;               ws += 64 * 64 * 2;
    __bf16* Wr0p = (__bf16*)ws;               ws += 64 * 64 * 2;
    __bf16* Wl1p = (__bf16*)ws;               ws += 64 * 32 * 2;
    __bf16* Wr1p = (__bf16*)ws;               ws += 64 * 32 * 2;
    __bf16* Wl2p = (__bf16*)ws;               ws += 32 * 16 * 2;
    __bf16* Wr2p = (__bf16*)ws;               ws += 32 * 16 * 2;

    // --- degree counts (once) ---
    zero_kernel<<<1024, 256, 0, stream>>>(cnt, (long)N_NODES);
    count_kernel<<<(N_EDGES + 255) / 256, 256, 0, stream>>>(dst, cnt, N_EDGES);

    // --- pack weights into WMMA B-operand layout ---
    pack_weights_kernel<<<(64 * 64 + 255) / 256, 256, 0, stream>>>(Wl0, Wl0p, 64, 64);
    pack_weights_kernel<<<(64 * 64 + 255) / 256, 256, 0, stream>>>(Wr0, Wr0p, 64, 64);
    pack_weights_kernel<<<(64 * 32 + 255) / 256, 256, 0, stream>>>(Wl1, Wl1p, 64, 32);
    pack_weights_kernel<<<(64 * 32 + 255) / 256, 256, 0, stream>>>(Wr1, Wr1p, 64, 32);
    pack_weights_kernel<<<(32 * 16 + 255) / 256, 256, 0, stream>>>(Wl2, Wl2p, 32, 16);
    pack_weights_kernel<<<(32 * 16 + 255) / 256, 256, 0, stream>>>(Wr2, Wr2p, 32, 16);

    // --- Layer 0: 64 -> 64 ---
    {
        long total = (long)N_EDGES * (64 / 4);
        zero_kernel<<<4096, 256, 0, stream>>>(s, (long)N_NODES * 64);
        edge_scatter_kernel<<<(int)((total + 255) / 256), 256, 0, stream>>>(
            x, src, dst, s, total, 64 / 4, 64);
        dim3 g(N_NODES / 16, 64 / 16);
        sage_gemm_kernel<<<g, 32, 0, stream>>>(s, cnt, x, Wl0p, Wr0p, b0, h0, 64, 64, 4);
    }
    // --- Layer 1: 64 -> 32 ---
    {
        long total = (long)N_EDGES * (64 / 4);
        zero_kernel<<<4096, 256, 0, stream>>>(s, (long)N_NODES * 64);
        edge_scatter_kernel<<<(int)((total + 255) / 256), 256, 0, stream>>>(
            h0, src, dst, s, total, 64 / 4, 64);
        dim3 g(N_NODES / 16, 32 / 16);
        sage_gemm_kernel<<<g, 32, 0, stream>>>(s, cnt, h0, Wl1p, Wr1p, b1, h1, 64, 32, 2);
    }
    // --- Layer 2: 32 -> 16 ---
    {
        long total = (long)N_EDGES * (32 / 4);
        zero_kernel<<<4096, 256, 0, stream>>>(s, (long)N_NODES * 32);
        edge_scatter_kernel<<<(int)((total + 255) / 256), 256, 0, stream>>>(
            h1, src, dst, s, total, 32 / 4, 32);
        dim3 g(N_NODES / 16, 16 / 16);
        sage_gemm_kernel<<<g, 32, 0, stream>>>(s, cnt, h1, Wl2p, Wr2p, b2, h2, 32, 16, 1);
    }
    // --- Head: 16 -> 1 ---
    final_head_kernel<<<(N_NODES + 255) / 256, 256, 0, stream>>>(h2, Wc, bc, (float*)d_out, N_NODES);
}
